// EdgeModel_22840636080545
// MI455X (gfx1250) — matrix operand order repaired and checked by reading, not compiled, not run on hardware
//
#include <hip/hip_runtime.h>
#include <hip/hip_bf16.h>

typedef __attribute__((ext_vector_type(16))) _Float16 v16h;
typedef __attribute__((ext_vector_type(8)))  _Float16 v8h;
typedef __attribute__((ext_vector_type(8)))  float    v8f;
typedef __attribute__((ext_vector_type(4)))  unsigned int v4u;
typedef __attribute__((ext_vector_type(8)))  int      v8i;
typedef __attribute__((ext_vector_type(4)))  int      v4i;

#define CC   512
#define NS   384     // N_STEP == N_STATE
#define NROWS 768    // step rows ++ state rows

// ---------------- workspace layout (bytes) ----------------
// 0        : preT16  3*512*512 f16  | 1572864 : w1T16 | 2097152 : w2T16
// 2621440  : bufX 768*512 f16       | 3407872 : bufY  | 4194304 : uv16
// 4980736  : uv32 768*512 f32       | 6553600 : matchws 384*384 f32
// 7143424  : argmaxws 384 i32

// ---------------- output layout (floats) ----------------
// 0       ss_logits [384,384,2] | 294912 st_logits | 589824 ss_scores
// 737280  st_edge_scores | 884736 matched | 1032192 match_mask

__device__ __forceinline__ v16h pack16(v8h lo, v8h hi8) {
    v16h a;
#pragma unroll
    for (int e = 0; e < 8; ++e) { a[e] = lo[e]; a[8 + e] = hi8[e]; }
    return a;
}

// A fragment (16x32 f16, MxK): lane row = lane&15; element e -> K = 8*hi + (e&7) + 16*(e>>3)
__device__ __forceinline__ v16h load_a_frag(const _Float16* __restrict__ rowp, int kb, int hi) {
    v8h lo = *(const v8h*)(rowp + kb + 8 * hi);
    v8h hh = *(const v8h*)(rowp + kb + 16 + 8 * hi);
    return pack16(lo, hh);
}

// B fragment (32x16 f16, KxN) from transposed weights [N][K]: lane col = lane&15; e -> K = 8*hi + (e&7) + 16*(e>>3)
__device__ __forceinline__ v16h load_b_frag(const _Float16* __restrict__ colp, int kb, int hi) {
    v8h lo = *(const v8h*)(colp + kb + 8 * hi);
    v8h hh = *(const v8h*)(colp + kb + 16 + 8 * hi);
    return pack16(lo, hh);
}

__device__ __forceinline__ v8f wmma16(v16h a, v16h b, v8f c) {
    return __builtin_amdgcn_wmma_f32_16x16x32_f16(false, a, false, b, (short)0, c, false, false);
}

// act1 fragment on the fly: relu(partner_j + (u_i + b1)) via packed-f16 ops
// (vector + -> v_pk_add_f16; elementwise_max(maxnum) -> v_pk_max_num_f16)
__device__ __forceinline__ v16h act1_frag(const _Float16* __restrict__ prow,
                                          const _Float16* __restrict__ wrow, int kb, int hi) {
    const int o0 = kb + 8 * hi, o1 = kb + 16 + 8 * hi;
    const v8h z = {};
    v8h a = *(const v8h*)(prow + o0) + *(const v8h*)(wrow + o0);
    v8h b = *(const v8h*)(prow + o1) + *(const v8h*)(wrow + o1);
    a = __builtin_elementwise_max(a, z);
    b = __builtin_elementwise_max(b, z);
    return pack16(a, b);
}

__device__ __forceinline__ float sigm(float x) { return 1.f / (1.f + __expf(-x)); }

// ---- Tensor Data Mover: DMA a 2D slab W2T[n0..n0+127][kb..kb+31] (f16) into LDS.
// D# per CDNA5 ISA ch.8: data_size=2B, tile 32x128, tensor_dim0_stride=512,
// pad_enable: interval 16 DWORDs (one 32-half row), amount 4 DWORDs (8 halves)
// -> LDS row stride = 40 halves (bank-spread), matching the compute-side layout.
__device__ __forceinline__ void tdm_load_slab(const _Float16* gsrc, unsigned lds_byte_addr) {
    unsigned long long ga = (unsigned long long)(uintptr_t)gsrc;
    v4u g0;
    g0.x = 1u;                                   // count=1 valid descriptor
    g0.y = lds_byte_addr;                        // LDS dest (bytes)
    g0.z = (unsigned)(ga & 0xffffffffu);         // global addr [31:0]
    g0.w = (unsigned)((ga >> 32) & 0x01ffffffu)  // global addr [56:32]
         | (2u << 30);                           // type = 2 ("image")
    v8i g1;
    g1[0] = (int)((1u << 16)      // data_size = 1 -> 2 bytes
                | (1u << 20)      // pad_enable
                | (3u << 22)      // pad_interval: 16 DWORDs
                | (3u << 25));    // pad_amount : 4 DWORDs
    g1[1] = (int)(512u << 16);    // tensor_dim0 = 512 (low16 -> bits 63:48)
    g1[2] = (int)(512u << 16);    // tensor_dim0 hi=0 | tensor_dim1 = 512 low16
    g1[3] = (int)(32u << 16);     // tensor_dim1 hi=0 | tile_dim0 = 32
    g1[4] = 128;                  // tile_dim1 = 128, tile_dim2 = 0
    g1[5] = 512;                  // tensor_dim0_stride = 512 (low 32)
    g1[6] = 0;                    // stride hi / dim1_stride lo
    g1[7] = 0;
    v4i z4 = {0, 0, 0, 0};
    v8i z8 = {0, 0, 0, 0, 0, 0, 0, 0};
    __builtin_amdgcn_tensor_load_to_lds(g0, g1, z4, z4, z8, 0);
}

// ---------------- fused pairwise kernel (HOT LOOP — kept first for disasm) ----
// WG = 256 thr (8 waves): one blk, two i values, 128 j values. Wave: 32 pair-rows.
// Layer1 fused via u/v linearity trick; layer2 WMMA with TDM double-buffered
// W2^T slabs in LDS (single barrier per stage); layer3 (512->2) fused as VALU
// dot + lane butterfly.
__global__ __launch_bounds__(256) void pairwise_kernel(
    const _Float16* __restrict__ P16,  // uv16 [768][512]
    const float*    __restrict__ U32,  // uv32 (step rows)
    const float*    __restrict__ B1,
    const _Float16* __restrict__ W2T,
    const float*    __restrict__ B2,
    const float*    __restrict__ W3,   // [512][2]
    const float*    __restrict__ B3,   // [2]
    float* __restrict__ out, float* __restrict__ matchws)
{
    __shared__ alignas(16) _Float16 s_w[2][CC];          // u_i + b1 (per i)
    __shared__ alignas(16) _Float16 s_b[2][128 * 40];    // double-buffered W2^T slabs

    const int tid = threadIdx.x, wave = tid >> 5, lane = tid & 31;
    const int hi = lane >> 4, l15 = lane & 15;
    const int wg = blockIdx.x;
    const int blk = wg / 576;          // 0 = step-step, 1 = step-state
    const int rem = wg % 576;
    const int i0 = (rem / 3) * 2, jg = rem % 3;

#pragma unroll
    for (int q = 0; q < 4; ++q) {      // fill s_w: 1024 entries / 256 threads
        const int idx = tid + 256 * q;
        const int is = idx >> 9, kk = idx & 511;
        s_w[is][kk] = (_Float16)(U32[(i0 + is) * CC + kk] + B1[kk]);
    }

    const int i = i0 + (wave >> 2);
    const int j0w = jg * 128 + (wave & 3) * 32;
    const _Float16* pt  = P16 + (blk ? NS * CC : 0);
    const _Float16* pr0 = pt + (j0w + l15) * CC;
    const _Float16* pr1 = pr0 + 16 * CC;
    const _Float16* wrow = s_w[wave >> 2];

    // TDM pipeline: stage s = nt*16 + k loads W2T[nt*128..+127][k*32..+31] -> s_b[s&1]
    if (wave == 0) {
        tdm_load_slab(W2T, (unsigned)(uintptr_t)&s_b[0][0]);   // stage 0
    }
    __syncthreads();   // publish s_w (act1 below reads it pre-barrier each stage)

    float p3[2][8][2];
#pragma unroll
    for (int g = 0; g < 2; ++g)
#pragma unroll
        for (int r = 0; r < 8; ++r) { p3[g][r][0] = 0.f; p3[g][r][1] = 0.f; }

    for (int nt = 0; nt < 4; ++nt) {
        v8f acc[2][8];
#pragma unroll
        for (int g = 0; g < 2; ++g)
#pragma unroll
            for (int t = 0; t < 8; ++t) acc[g][t] = v8f{};

        for (int k = 0; k < 16; ++k) {
            const int s = nt * 16 + k;
            const int kb = k * 32;
            // A fragments are slab-independent: issue their loads BEFORE the
            // tensor-wait + barrier so they fly while we synchronize.
            __builtin_prefetch(pr0 + kb + 32, 0, 1);    // next-k partner segment
            v16h a0 = act1_frag(pr0, wrow, kb, hi);
            v16h a1 = act1_frag(pr1, wrow, kb, hi);
            if (wave == 0)
                __builtin_amdgcn_s_wait_tensorcnt(0);   // slab s landed
            // Single barrier per stage: publishes slab s AND certifies that all
            // waves retired slab s-1 (so the TDM below may overwrite its buffer).
            __syncthreads();
            if (wave == 0 && s + 1 < 64) {
                const int s1 = s + 1;
                const _Float16* gsrc = W2T + ((s1 >> 4) * 128) * CC + (s1 & 15) * 32;
                tdm_load_slab(gsrc, (unsigned)(uintptr_t)&s_b[s1 & 1][0]);  // overlap with compute
            }
            const _Float16* sb = s_b[s & 1];
#pragma unroll
            for (int hb = 0; hb < 2; ++hb) {            // 2 batches: 8 ds loads then 8 WMMAs
                v16h bfr[4];
#pragma unroll
                for (int t = 0; t < 4; ++t) {
                    const _Float16* bp = sb + ((hb * 4 + t) * 16 + l15) * 40 + hi * 8;
                    bfr[t] = pack16(*(const v8h*)bp, *(const v8h*)(bp + 16));
                }
#pragma unroll
                for (int t = 0; t < 4; ++t) {
                    acc[0][hb * 4 + t] = wmma16(a0, bfr[t], acc[0][hb * 4 + t]);
                    acc[1][hb * 4 + t] = wmma16(a1, bfr[t], acc[1][hb * 4 + t]);
                }
            }
        }
        // bias + relu + fused layer3 partial dot (act2 never leaves registers)
#pragma unroll
        for (int t = 0; t < 8; ++t) {
            const int col = nt * 128 + t * 16 + l15;
            const float bb = B2[col];
            const float w30 = W3[col * 2], w31 = W3[col * 2 + 1];
#pragma unroll
            for (int r = 0; r < 8; ++r) {
                const float v0 = fmaxf(acc[0][t][r] + bb, 0.f);
                const float v1 = fmaxf(acc[1][t][r] + bb, 0.f);
                p3[0][r][0] += v0 * w30; p3[0][r][1] += v0 * w31;
                p3[1][r][0] += v1 * w30; p3[1][r][1] += v1 * w31;
            }
        }
    }

    // butterfly reduce over the 16 lanes (l15) sharing each output row
#pragma unroll
    for (int m = 1; m < 16; m <<= 1) {
#pragma unroll
        for (int g = 0; g < 2; ++g)
#pragma unroll
            for (int r = 0; r < 8; ++r) {
                p3[g][r][0] += __shfl_xor(p3[g][r][0], m, 32);
                p3[g][r][1] += __shfl_xor(p3[g][r][1], m, 32);
            }
    }

    if (l15 == 0) {   // lanes 0 and 16 active -> rows r+0 and r+8 per store
        const float b30 = B3[0], b31 = B3[1];
        float* logits = out + (blk ? 294912 : 0);
        float* scores = out + (blk ? 737280 : 589824);
#pragma unroll
    for (int g = 0; g < 2; ++g)
#pragma unroll
            for (int r = 0; r < 8; ++r) {
                const int j = j0w + g * 16 + r + 8 * hi;
                const int pr = i * NS + j;
                const float L0 = p3[g][r][0] + b30;
                const float L1 = p3[g][r][1] + b31;
                *(float2*)(logits + 2 * pr) = make_float2(L0, L1);
                const float s1 = sigm(L1);
                scores[pr] = (s1 > 0.05f) ? s1 : 0.f;
                if (blk) matchws[pr] = sigm(L0);
            }
    }
}

// ---------------- prep: convert/transpose weights + inputs to f16 ----------------
__global__ __launch_bounds__(256) void prep_kernel(
    const float* __restrict__ step_x, const float* __restrict__ state_x,
    const float* __restrict__ pre_w, const float* __restrict__ w1, const float* __restrict__ w2,
    _Float16* __restrict__ preT, _Float16* __restrict__ w1T, _Float16* __restrict__ w2T,
    _Float16* __restrict__ x16)
{
    int idx = blockIdx.x * 256 + threadIdx.x;
    switch (blockIdx.y) {
    case 0:
        if (idx < 3 * CC * CC) {
            int l = idx / (CC * CC), rem = idx % (CC * CC);
            int n = rem / CC, k = rem % CC;
            preT[idx] = (_Float16)pre_w[l * CC * CC + k * CC + n];
        }
        break;
    case 1:
        if (idx < CC * CC) { int n = idx / CC, k = idx % CC; w1T[idx] = (_Float16)w1[k * CC + n]; }
        break;
    case 2:
        if (idx < CC * CC) { int n = idx / CC, k = idx % CC; w2T[idx] = (_Float16)w2[k * CC + n]; }
        break;
    case 3:
        if (idx < NROWS * CC)
            x16[idx] = (_Float16)((idx < NS * CC) ? step_x[idx] : state_x[idx - NS * CC]);
        break;
    }
}

// ---------------- WMMA GEMM: Y[768,512] = act(X @ Wt^T + b) ----------------
// 8 waves / WG; wave tile = 16 rows x 128 cols; grid = 24 WGs.
// Explicit 2-stage software pipeline over k.
__global__ __launch_bounds__(256) void gemm768_kernel(
    const _Float16* __restrict__ X, const _Float16* __restrict__ WT,
    const float* __restrict__ bias, _Float16* __restrict__ Y16,
    float* __restrict__ Y32, int do_relu)
{
    const int tid = threadIdx.x, wave = tid >> 5, lane = tid & 31;
    const int hi = lane >> 4, l15 = lane & 15;
    const int tile = blockIdx.x * 8 + wave;
    const int M0 = (tile >> 2) * 16, Nb = (tile & 3) * 128;
    const _Float16* arow = X + (M0 + l15) * CC;

    v8f acc[8];
#pragma unroll
    for (int t = 0; t < 8; ++t) acc[t] = v8f{};

    v16h a0 = load_a_frag(arow, 0, hi);
    v16h b0[8];
#pragma unroll
    for (int t = 0; t < 8; ++t)
        b0[t] = load_b_frag(WT + (Nb + t * 16 + l15) * CC, 0, hi);

    for (int k = 0; k < 16; k += 2) {
        const int kb1 = (k + 1) * 32;
        v16h a1 = load_a_frag(arow, kb1, hi);
        v16h b1[8];
#pragma unroll
        for (int t = 0; t < 8; ++t)
            b1[t] = load_b_frag(WT + (Nb + t * 16 + l15) * CC, kb1, hi);
#pragma unroll
        for (int t = 0; t < 8; ++t)
            acc[t] = wmma16(a0, b0[t], acc[t]);
        if (k + 2 < 16) {
            const int kb2 = (k + 2) * 32;
            a0 = load_a_frag(arow, kb2, hi);
#pragma unroll
            for (int t = 0; t < 8; ++t)
                b0[t] = load_b_frag(WT + (Nb + t * 16 + l15) * CC, kb2, hi);
        }
#pragma unroll
        for (int t = 0; t < 8; ++t)
            acc[t] = wmma16(a1, b1[t], acc[t]);
    }
#pragma unroll
    for (int t = 0; t < 8; ++t) {
        const int col = Nb + t * 16 + l15;
        const float bb = bias ? bias[col] : 0.f;
#pragma unroll
        for (int r = 0; r < 8; ++r) {
            const int row = M0 + r + 8 * hi;
            float v = acc[t][r] + bb;
            if (do_relu) v = fmaxf(v, 0.f);
            Y16[row * CC + col] = (_Float16)v;
            if (Y32) Y32[row * CC + col] = v;
        }
    }
}

// ---------------- mutual-argmax epilogue ----------------
__global__ void col_argmax_kernel(const float* __restrict__ match, int* __restrict__ am) {
    const int m = blockIdx.x * blockDim.x + threadIdx.x;
    if (m >= NS) return;
    float best = -3.4e38f; int bi = 0;
    for (int ii = 0; ii < NS; ++ii) {
        const float v = match[ii * NS + m];
        if (v > best) { best = v; bi = ii; }
    }
    am[m] = bi;
}

__global__ void row_finish_kernel(const float* __restrict__ match, const int* __restrict__ am,
                                  float* __restrict__ matched, float* __restrict__ mask) {
    const int ii = blockIdx.x * blockDim.x + threadIdx.x;
    if (ii >= NS) return;
    float best = -3.4e38f; int bj = 0;
    for (int j = 0; j < NS; ++j) {
        const float mv = (am[j] == ii) ? match[ii * NS + j] : 0.f;
        if (mv > best) { best = mv; bj = j; }
    }
    for (int j = 0; j < NS; ++j) {
        const float mv = (am[j] == ii) ? match[ii * NS + j] : 0.f;
        const float o = (j == bj) ? mv : 0.f;
        matched[ii * NS + j] = o;
        mask[ii * NS + j] = (o > 0.5f) ? 1.f : 0.f;
    }
}

extern "C" void kernel_launch(void* const* d_in, const int* in_sizes, int n_in,
                              void* d_out, int out_size, void* d_ws, size_t ws_size,
                              hipStream_t stream) {
    (void)in_sizes; (void)n_in; (void)out_size; (void)ws_size;
    const float* step_x  = (const float*)d_in[0];
    const float* state_x = (const float*)d_in[1];
    const float* pre_w   = (const float*)d_in[2];
    const float* pre_b   = (const float*)d_in[3];
    const float* w1      = (const float*)d_in[4];
    const float* b1      = (const float*)d_in[5];
    const float* w2      = (const float*)d_in[6];
    const float* b2      = (const float*)d_in[7];
    const float* w3      = (const float*)d_in[8];
    const float* b3      = (const float*)d_in[9];
    float* out = (float*)d_out;
    char*  ws  = (char*)d_ws;

    _Float16* preT = (_Float16*)(ws + 0);
    _Float16* w1T  = (_Float16*)(ws + 1572864);
    _Float16* w2T  = (_Float16*)(ws + 2097152);
    _Float16* bufX = (_Float16*)(ws + 2621440);
    _Float16* bufY = (_Float16*)(ws + 3407872);
    _Float16* uv16 = (_Float16*)(ws + 4194304);
    float*    uv32 = (float*)   (ws + 4980736);
    float*    mws  = (float*)   (ws + 6553600);
    int*      amws = (int*)     (ws + 7143424);

    prep_kernel<<<dim3(3072, 4), 256, 0, stream>>>(step_x, state_x, pre_w, w1, w2,
                                                   preT, w1T, w2T, bufX);
    // pre_compare: 3 layers (relu, relu, linear) on concat(step,state)
    gemm768_kernel<<<24, 256, 0, stream>>>(bufX, preT,               pre_b,          bufY, nullptr, 1);
    gemm768_kernel<<<24, 256, 0, stream>>>(bufY, preT + CC * CC,     pre_b + CC,     bufX, nullptr, 1);
    gemm768_kernel<<<24, 256, 0, stream>>>(bufX, preT + 2 * CC * CC, pre_b + 2 * CC, bufY, nullptr, 0);
    // u/v = h @ W1 (f16 partners + f32 for the i-side broadcast array)
    gemm768_kernel<<<24, 256, 0, stream>>>(bufY, w1T, nullptr, uv16, uv32, 0);
    // fused pairwise post_compare (both ss and st blocks)
    pairwise_kernel<<<1152, 256, 0, stream>>>(uv16, uv32, b1, w2T, b2, w3, b3, out, mws);
    // mutual argmax epilogue
    col_argmax_kernel<<<3, 128, 0, stream>>>(mws, amws);
    row_finish_kernel<<<3, 128, 0, stream>>>(mws, amws, out + 884736, out + 1032192);
}
